// ShuffleSelfAttentionLayer_89292370083972
// MI455X (gfx1250) — matrix operand order repaired and checked by reading, not compiled, not run on hardware
//
#include <hip/hip_runtime.h>
#include <hip/hip_bf16.h>

// ---------------------------------------------------------------------------
// Types for CDNA5 WMMA (wave32): D(f32 16x16) = A(bf16 16x32) * B(bf16 32x16) + C
// ---------------------------------------------------------------------------
typedef __bf16 bf16;
typedef __attribute__((ext_vector_type(16))) __bf16 v16bf;
typedef __attribute__((ext_vector_type(8)))  float  v8f;

static __device__ __forceinline__ v8f wmma_bf16(v16bf a, v16bf b, v8f c) {
  return __builtin_amdgcn_wmma_f32_16x16x32_bf16(false, a, false, b, (short)0, c, false, false);
}

// A fragment (16x32 bf16): lane l holds row m0+(l&15);
// lanes 0-15: K = kb+0..15; lanes 16-31: K = kb+16..31.
static __device__ __forceinline__ v16bf ld_a(const bf16* __restrict__ A, int ld,
                                             int m0, int kb, int lane) {
  const bf16* p = A + (m0 + (lane & 15)) * ld + kb + ((lane >> 4) << 4);
  v16bf f;
#pragma unroll
  for (int i = 0; i < 16; ++i) f[i] = p[i];
  return f;
}

// B fragment (32x16 bf16) from a [N][K] ("transposed") matrix: lane l holds
// column n0+(l&15); lanes 0-15: K = kb+0..15; lanes 16-31: K = kb+16..31.
static __device__ __forceinline__ v16bf ld_bt(const bf16* __restrict__ Bt, int ld,
                                              int n0, int kb, int lane) {
  const bf16* p = Bt + (n0 + (lane & 15)) * ld + kb + ((lane >> 4) << 4);
  v16bf f;
#pragma unroll
  for (int i = 0; i < 16; ++i) f[i] = p[i];
  return f;
}

// ---------------------------------------------------------------------------
// CDNA5 async global->LDS copy (GLOBAL_LOAD_ASYNC_TO_LDS_B128, ASYNCcnt).
// Each lane copies 16 bytes from its global address to its LDS byte address.
// ---------------------------------------------------------------------------
static __device__ __forceinline__ void async_copy_b128(const bf16* gptr, bf16* lptr) {
  unsigned lds_off = (unsigned)(size_t)lptr;   // LDS aperture: low 32 bits = LDS byte addr
  asm volatile("global_load_async_to_lds_b128 %0, %1, off"
               :: "v"(lds_off), "v"(gptr)
               : "memory");
}
static __device__ __forceinline__ void wait_async0() {
  asm volatile("s_wait_asynccnt 0x0" ::: "memory");
}

// ---------------------------------------------------------------------------
// Problem constants
// ---------------------------------------------------------------------------
#define BATCH   64
#define DFEAT   512
#define NDEPTH  64
#define NHEAD   8
#define EWIDTH  4096          // N_CH * D_FEAT
#define DHID    2048
#define MROWS   4096          // BATCH * NDEPTH
#define DH      512           // EWIDTH / NHEAD (== DFEAT, so vchannel v == head h)

// ---------------------------------------------------------------------------
// 1) Feature shuffle-gather -> bf16 activation matrix fm[4096][512]
// ---------------------------------------------------------------------------
__global__ __launch_bounds__(256) void gather_kernel(
    const float* __restrict__ features, const int* __restrict__ shuffle_idx,
    bf16* __restrict__ fm_bf) {
  int idx = blockIdx.x * blockDim.x + threadIdx.x;   // over 4096*512
  if (idx >= MROWS * DFEAT) return;
  int f = idx & (DFEAT - 1);
  int m = idx >> 9;            // b*64 + n
  int b = m >> 6;
  int n = m & 63;
  fm_bf[idx] = (bf16)features[b * DFEAT + shuffle_idx[n * DFEAT + f]];
}

// ---------------------------------------------------------------------------
// 2) Weight transpose+convert: in f32 [K][N] (row-major) -> out bf16 [N][K]
// ---------------------------------------------------------------------------
__global__ __launch_bounds__(256) void transpose_w_kernel(
    const float* __restrict__ in, bf16* __restrict__ out, int K, int N) {
  int idx = blockIdx.x * blockDim.x + threadIdx.x;
  if (idx >= K * N) return;
  int n = idx % N;
  int k = idx / N;
  out[n * K + k] = (bf16)in[idx];
}

// ---------------------------------------------------------------------------
// 3) QKV projection GEMM: [4096x512] @ [512x4096] + bias, bf16 WMMA.
//    grid = (32, 32, 3); block = 256 (8 waves). Block tile 128x128,
//    wave tile 32x64 (wave grid 4 down x 2 across).
//    LDS double-buffered K-tiles staged with async global->LDS copies so the
//    next tile streams in (ASYNCcnt) while WMMAs consume the current one.
//    z==2 (V) is stored transposed Vt[e][m] so the P@V GEMM streams K.
// ---------------------------------------------------------------------------
__global__ __launch_bounds__(256) void qkv_gemm_kernel(
    const bf16* __restrict__ fm, const bf16* __restrict__ Wqt,
    const bf16* __restrict__ Wkt, const bf16* __restrict__ Wvt,
    const float* __restrict__ bq, const float* __restrict__ bk,
    const float* __restrict__ bv,
    bf16* __restrict__ Qbf, bf16* __restrict__ Kbf, bf16* __restrict__ Vt) {
  __shared__ __align__(16) bf16 Atile[2][128][32];   // [m][k], 8KB each
  __shared__ __align__(16) bf16 Btile[2][128][32];   // [n][k], 8KB each

  const int tid  = threadIdx.x;
  const int lane = tid & 31;
  const int wave = tid >> 5;
  const int wm = (wave & 3) * 32;      // wave row offset inside block tile
  const int wn = (wave >> 2) * 64;     // wave col offset inside block tile
  const int bm = blockIdx.x * 128;
  const int bn = blockIdx.y * 128;
  const int z  = blockIdx.z;
  const bf16*  Wt   = (z == 0) ? Wqt : (z == 1) ? Wkt : Wvt;
  const float* bias = (z == 0) ? bq  : (z == 1) ? bk  : bv;

  const bf16* Ag = fm + (size_t)bm * DFEAT;   // A rows for this block
  const bf16* Bg = Wt + (size_t)bn * DFEAT;   // B (transposed) rows for this block

  // Stage one 128x32 K-tile of A and B into LDS buffer `buf` (async).
  // 2 x 16B chunks per thread per matrix; chunk c -> row = c>>2, koff = (c&3)*8.
  auto stage = [&](int buf, int kb) {
#pragma unroll
    for (int i = 0; i < 2; ++i) {
      int c   = tid + (i << 8);
      int row = c >> 2;
      int ko  = (c & 3) << 3;
      async_copy_b128(Ag + (size_t)row * DFEAT + kb + ko, &Atile[buf][row][ko]);
      async_copy_b128(Bg + (size_t)row * DFEAT + kb + ko, &Btile[buf][row][ko]);
    }
  };

  v8f acc[2][4];
#pragma unroll
  for (int s = 0; s < 2; ++s)
#pragma unroll
    for (int t = 0; t < 4; ++t) acc[s][t] = (v8f)0.0f;

  stage(0, 0);
  wait_async0();
  __syncthreads();

  const int KSTEPS = DFEAT / 32;   // 16
#pragma unroll 1
  for (int ks = 0; ks < KSTEPS; ++ks) {
    const int cur = ks & 1;
    if (ks + 1 < KSTEPS) stage(cur ^ 1, (ks + 1) * 32);   // prefetch next tile

    const bf16* At = &Atile[cur][0][0];
    const bf16* Bt = &Btile[cur][0][0];
    v16bf a0 = ld_a(At, 32, wm,      0, lane);
    v16bf a1 = ld_a(At, 32, wm + 16, 0, lane);
#pragma unroll
    for (int t = 0; t < 4; ++t) {
      v16bf bb = ld_bt(Bt, 32, wn + t * 16, 0, lane);
      acc[0][t] = wmma_bf16(a0, bb, acc[0][t]);
      acc[1][t] = wmma_bf16(a1, bb, acc[1][t]);
    }

    if (ks + 1 < KSTEPS) {
      wait_async0();       // own async copies landed
      __syncthreads();     // everyone's copies landed; reads of `cur` done
    }
  }

  // Epilogue: bias add, convert, store (V transposed).
#pragma unroll
  for (int s = 0; s < 2; ++s) {
#pragma unroll
    for (int t = 0; t < 4; ++t) {
      int n = bn + wn + t * 16 + (lane & 15);
      float bb = bias[n];
#pragma unroll
      for (int r = 0; r < 8; ++r) {
        int m = bm + wm + s * 16 + r + ((lane >> 4) << 3);
        float val = acc[s][t][r] + bb;
        if (z == 0)      Qbf[(size_t)m * EWIDTH + n] = (bf16)val;
        else if (z == 1) Kbf[(size_t)m * EWIDTH + n] = (bf16)val;
        else             Vt [(size_t)n * MROWS  + m] = (bf16)val;
      }
    }
  }
}

// ---------------------------------------------------------------------------
// 4) Attention per (head, batch): S = Q*K^T/8 -> softmax -> O = P*V.
//    grid = 512 (h*64+b); block = 128 (4 waves).
// ---------------------------------------------------------------------------
__global__ __launch_bounds__(128) void attn_kernel(
    const bf16* __restrict__ Qbf, const bf16* __restrict__ Kbf,
    const bf16* __restrict__ Vt,
    float* __restrict__ attn_probs,       // [H*B, 64, 64] f32 (part of d_out)
    float* __restrict__ attnout) {        // [H, B, 64, 512] f32
  __shared__ float S[NDEPTH][NDEPTH];
  __shared__ __align__(16) bf16 P[NDEPTH][NDEPTH];

  const int h = blockIdx.x >> 6;
  const int b = blockIdx.x & 63;
  const int lane = threadIdx.x & 31;
  const int wave = threadIdx.x >> 5;
  const int m0 = wave * 16;

  const bf16* Qhb = Qbf + (size_t)(b * NDEPTH) * EWIDTH + h * DH;
  const bf16* Khb = Kbf + (size_t)(b * NDEPTH) * EWIDTH + h * DH;  // B^T view

  // ---- S = Q @ K^T  (M=64, N=64, K=512) ----
  v8f acc[4];
#pragma unroll
  for (int t = 0; t < 4; ++t) acc[t] = (v8f)0.0f;
  for (int kb = 0; kb < DH; kb += 32) {
    v16bf a = ld_a(Qhb, EWIDTH, m0, kb, lane);
#pragma unroll
    for (int t = 0; t < 4; ++t) {
      v16bf bb = ld_bt(Khb, EWIDTH, t * 16, kb, lane);
      acc[t] = wmma_bf16(a, bb, acc[t]);
    }
  }
  const float scale = 0.125f;   // 1/sqrt(N_DEPTH) = 1/8
#pragma unroll
  for (int t = 0; t < 4; ++t)
#pragma unroll
    for (int r = 0; r < 8; ++r)
      S[m0 + r + ((lane >> 4) << 3)][t * 16 + (lane & 15)] = acc[t][r] * scale;
  __syncthreads();

  // ---- row softmax (threads 0..63, one row each) ----
  if (threadIdx.x < NDEPTH) {
    int q = threadIdx.x;
    float mx = -3.4e38f;
    for (int k = 0; k < NDEPTH; ++k) mx = fmaxf(mx, S[q][k]);
    float sum = 0.0f;
    for (int k = 0; k < NDEPTH; ++k) { float e = __expf(S[q][k] - mx); S[q][k] = e; sum += e; }
    float inv = 1.0f / sum;
    float* dst = attn_probs + ((size_t)(h * BATCH + b) * NDEPTH + q) * NDEPTH;
    for (int k = 0; k < NDEPTH; ++k) {
      float p = S[q][k] * inv;
      dst[k]  = p;
      P[q][k] = (bf16)p;
    }
  }
  __syncthreads();

  // ---- O = P @ V  (M=64, N=512, K=64), V held transposed Vt[e][m] ----
  const bf16* Vhb = Vt + (size_t)(h * DH) * MROWS + b * NDEPTH;  // row n, col = b*64+k
  float* Ohb = attnout + (size_t)(h * BATCH + b) * NDEPTH * DH;
  for (int nc = 0; nc < DH; nc += 64) {
    v8f o[4];
#pragma unroll
    for (int t = 0; t < 4; ++t) o[t] = (v8f)0.0f;
#pragma unroll
    for (int kb = 0; kb < NDEPTH; kb += 32) {
      v16bf a = ld_a(&P[0][0], NDEPTH, m0, kb, lane);
#pragma unroll
      for (int t = 0; t < 4; ++t) {
        v16bf bb = ld_bt(Vhb, MROWS, nc + t * 16, kb, lane);
        o[t] = wmma_bf16(a, bb, o[t]);
      }
    }
#pragma unroll
    for (int t = 0; t < 4; ++t)
#pragma unroll
      for (int r = 0; r < 8; ++r)
        Ohb[(size_t)(m0 + r + ((lane >> 4) << 3)) * DH + nc + t * 16 + (lane & 15)] = o[t][r];
  }
}

// ---------------------------------------------------------------------------
// 5) conv (weighted sum over h,n) + residual + LayerNorm1. grid=64, block=512.
// ---------------------------------------------------------------------------
__global__ __launch_bounds__(512) void conv_ln1_kernel(
    const float* __restrict__ attnout, const float* __restrict__ conv_w,
    const float* __restrict__ features,
    const float* __restrict__ g, const float* __restrict__ beta,
    float* __restrict__ xf, bf16* __restrict__ xbf) {
  __shared__ float cw[NHEAD * NDEPTH];
  __shared__ float s1[DFEAT], s2[DFEAT];
  const int b = blockIdx.x;
  const int f = threadIdx.x;
  cw[f] = conv_w[f];
  __syncthreads();

  float s = 0.0f;
#pragma unroll 1
  for (int h = 0; h < NHEAD; ++h) {
    const float* base = attnout + (size_t)(h * BATCH + b) * NDEPTH * DH + f;
    const float* w    = cw + h * NDEPTH;
    for (int n = 0; n < NDEPTH; ++n) s += base[(size_t)n * DH] * w[n];
  }
  float t = s + features[b * DFEAT + f];

  s1[f] = t; s2[f] = t * t;
  __syncthreads();
  for (int st = DFEAT / 2; st > 0; st >>= 1) {
    if (f < st) { s1[f] += s1[f + st]; s2[f] += s2[f + st]; }
    __syncthreads();
  }
  float mu  = s1[0] * (1.0f / DFEAT);
  float var = s2[0] * (1.0f / DFEAT) - mu * mu;
  float x = (t - mu) * rsqrtf(var + 1e-5f) * g[f] + beta[f];
  xf [b * DFEAT + f] = x;
  xbf[b * DFEAT + f] = (bf16)x;
}

// ---------------------------------------------------------------------------
// 6) MLP layer 1: H = relu(x @ W1 + b1). M=64,K=512,N=2048. grid=32, block=128.
// ---------------------------------------------------------------------------
__global__ __launch_bounds__(128) void mlp1_kernel(
    const bf16* __restrict__ xbf, const bf16* __restrict__ W1t,
    const float* __restrict__ b1, bf16* __restrict__ Hbf) {
  const int lane = threadIdx.x & 31;
  const int wave = threadIdx.x >> 5;
  const int m0 = wave * 16;
  const int n0 = blockIdx.x * 64;
  v8f acc[4];
#pragma unroll
  for (int t = 0; t < 4; ++t) acc[t] = (v8f)0.0f;
  for (int kb = 0; kb < DFEAT; kb += 32) {
    v16bf a = ld_a(xbf, DFEAT, m0, kb, lane);
#pragma unroll
    for (int t = 0; t < 4; ++t) {
      v16bf bb = ld_bt(W1t, DFEAT, n0 + t * 16, kb, lane);
      acc[t] = wmma_bf16(a, bb, acc[t]);
    }
  }
#pragma unroll
  for (int t = 0; t < 4; ++t) {
    int n = n0 + t * 16 + (lane & 15);
    float bias = b1[n];
#pragma unroll
    for (int r = 0; r < 8; ++r) {
      int m = m0 + r + ((lane >> 4) << 3);
      float v = acc[t][r] + bias;
      Hbf[m * DHID + n] = (bf16)(v > 0.0f ? v : 0.0f);
    }
  }
}

// ---------------------------------------------------------------------------
// 7) MLP layer 2: ypre = H @ W2 + b2. M=64,K=2048,N=512. grid=8, block=128.
// ---------------------------------------------------------------------------
__global__ __launch_bounds__(128) void mlp2_kernel(
    const bf16* __restrict__ Hbf, const bf16* __restrict__ W2t,
    const float* __restrict__ b2, float* __restrict__ ypre) {
  const int lane = threadIdx.x & 31;
  const int wave = threadIdx.x >> 5;
  const int m0 = wave * 16;
  const int n0 = blockIdx.x * 64;
  v8f acc[4];
#pragma unroll
  for (int t = 0; t < 4; ++t) acc[t] = (v8f)0.0f;
  for (int kb = 0; kb < DHID; kb += 32) {
    v16bf a = ld_a(Hbf, DHID, m0, kb, lane);
#pragma unroll
    for (int t = 0; t < 4; ++t) {
      v16bf bb = ld_bt(W2t, DHID, n0 + t * 16, kb, lane);
      acc[t] = wmma_bf16(a, bb, acc[t]);
    }
  }
#pragma unroll
  for (int t = 0; t < 4; ++t) {
    int n = n0 + t * 16 + (lane & 15);
    float bias = b2[n];
#pragma unroll
    for (int r = 0; r < 8; ++r) {
      int m = m0 + r + ((lane >> 4) << 3);
      ypre[m * DFEAT + n] = acc[t][r] + bias;
    }
  }
}

// ---------------------------------------------------------------------------
// 8) residual + LayerNorm2 -> final y into d_out. grid=64, block=512.
// ---------------------------------------------------------------------------
__global__ __launch_bounds__(512) void ln2_kernel(
    const float* __restrict__ ypre, const float* __restrict__ xf,
    const float* __restrict__ g, const float* __restrict__ beta,
    float* __restrict__ yout) {
  __shared__ float s1[DFEAT], s2[DFEAT];
  const int b = blockIdx.x;
  const int f = threadIdx.x;
  float t = ypre[b * DFEAT + f] + xf[b * DFEAT + f];
  s1[f] = t; s2[f] = t * t;
  __syncthreads();
  for (int st = DFEAT / 2; st > 0; st >>= 1) {
    if (f < st) { s1[f] += s1[f + st]; s2[f] += s2[f + st]; }
    __syncthreads();
  }
  float mu  = s1[0] * (1.0f / DFEAT);
  float var = s2[0] * (1.0f / DFEAT) - mu * mu;
  yout[b * DFEAT + f] = (t - mu) * rsqrtf(var + 1e-5f) * g[f] + beta[f];
}

// ---------------------------------------------------------------------------
// Host orchestration
// ---------------------------------------------------------------------------
extern "C" void kernel_launch(void* const* d_in, const int* in_sizes, int n_in,
                              void* d_out, int out_size, void* d_ws, size_t ws_size,
                              hipStream_t stream) {
  (void)in_sizes; (void)n_in; (void)out_size; (void)ws_size;
  const float* features = (const float*)d_in[0];
  const int*   shuf     = (const int*)  d_in[1];
  const float* Wq = (const float*)d_in[2];  const float* bq = (const float*)d_in[3];
  const float* Wk = (const float*)d_in[4];  const float* bk = (const float*)d_in[5];
  const float* Wv = (const float*)d_in[6];  const float* bv = (const float*)d_in[7];
  const float* conv_w = (const float*)d_in[8];
  const float* ln1_g = (const float*)d_in[9];  const float* ln1_b = (const float*)d_in[10];
  const float* W1 = (const float*)d_in[11]; const float* b1 = (const float*)d_in[12];
  const float* W2 = (const float*)d_in[13]; const float* b2 = (const float*)d_in[14];
  const float* ln2_g = (const float*)d_in[15]; const float* ln2_b = (const float*)d_in[16];

  float* y_out    = (float*)d_out;                       // [64, 512]
  float* attn_out = (float*)d_out + BATCH * DFEAT;       // [512, 64, 64]

  // Carve workspace
  char* w = (char*)d_ws;
  auto alloc = [&](size_t bytes) -> void* {
    void* p = (void*)w;
    w += (bytes + 255) & ~(size_t)255;
    return p;
  };
  bf16*  fm_bf = (bf16*) alloc((size_t)MROWS * DFEAT * 2);
  bf16*  Wqt   = (bf16*) alloc((size_t)EWIDTH * DFEAT * 2);
  bf16*  Wkt   = (bf16*) alloc((size_t)EWIDTH * DFEAT * 2);
  bf16*  Wvt   = (bf16*) alloc((size_t)EWIDTH * DFEAT * 2);
  bf16*  W1t   = (bf16*) alloc((size_t)DHID * DFEAT * 2);
  bf16*  W2t   = (bf16*) alloc((size_t)DFEAT * DHID * 2);
  bf16*  Qbf   = (bf16*) alloc((size_t)MROWS * EWIDTH * 2);
  bf16*  Kbf   = (bf16*) alloc((size_t)MROWS * EWIDTH * 2);
  bf16*  Vt    = (bf16*) alloc((size_t)EWIDTH * MROWS * 2);
  float* aout  = (float*)alloc((size_t)NHEAD * BATCH * NDEPTH * DH * 4);
  float* xf    = (float*)alloc((size_t)BATCH * DFEAT * 4);
  bf16*  xbf   = (bf16*) alloc((size_t)BATCH * DFEAT * 2);
  bf16*  Hbf   = (bf16*) alloc((size_t)BATCH * DHID * 2);
  float* ypre  = (float*)alloc((size_t)BATCH * DFEAT * 4);

  // 1) shuffle-gather to bf16
  gather_kernel<<<(MROWS * DFEAT) / 256, 256, 0, stream>>>(features, shuf, fm_bf);

  // 2) weight transposes to [N][K] bf16
  transpose_w_kernel<<<(DFEAT * EWIDTH) / 256, 256, 0, stream>>>(Wq, Wqt, DFEAT, EWIDTH);
  transpose_w_kernel<<<(DFEAT * EWIDTH) / 256, 256, 0, stream>>>(Wk, Wkt, DFEAT, EWIDTH);
  transpose_w_kernel<<<(DFEAT * EWIDTH) / 256, 256, 0, stream>>>(Wv, Wvt, DFEAT, EWIDTH);
  transpose_w_kernel<<<(DFEAT * DHID)  / 256, 256, 0, stream>>>(W1, W1t, DFEAT, DHID);
  transpose_w_kernel<<<(DHID * DFEAT)  / 256, 256, 0, stream>>>(W2, W2t, DHID, DFEAT);

  // 3) QKV projections (WMMA + async double-buffered LDS staging)
  qkv_gemm_kernel<<<dim3(MROWS / 128, EWIDTH / 128, 3), 256, 0, stream>>>(
      fm_bf, Wqt, Wkt, Wvt, bq, bk, bv, Qbf, Kbf, Vt);

  // 4) attention (WMMA + softmax), writes attn probs directly to d_out
  attn_kernel<<<NHEAD * BATCH, 128, 0, stream>>>(Qbf, Kbf, Vt, attn_out, aout);

  // 5) conv + residual + LN1
  conv_ln1_kernel<<<BATCH, DFEAT, 0, stream>>>(aout, conv_w, features, ln1_g, ln1_b, xf, xbf);

  // 6-7) MLP (WMMA)
  mlp1_kernel<<<DHID / 64, 128, 0, stream>>>(xbf, W1t, b1, Hbf);
  mlp2_kernel<<<DFEAT / 64, 128, 0, stream>>>(Hbf, W2t, b2, ypre);

  // 8) residual + LN2 -> y
  ln2_kernel<<<BATCH, DFEAT, 0, stream>>>(ypre, xf, ln2_g, ln2_b, y_out);
}
